// PAM_88210038326166
// MI455X (gfx1250) — compile-verified, matrix operand order checked by reading, not run on hardware
//
#include <hip/hip_runtime.h>
#include <hip/hip_bf16.h>

// ---------------------------------------------------------------------------
// PAM (position attention) for MI455X / gfx1250 — all GEMMs on the WMMA pipe.
//   stage 1: pack weights -> w16[640][512] f16, bias[640] f32
//   stage 2: transpose    -> xt[B][N][C] f16 via global_load_async_to_lds_b32
//   stage 3: projections  -> fb=Q[B][N][Cr], fc=K[B][N][Cr], fd=V[B][C][N]
//            2x4-tile blocked, fully-unrolled software-pipelined WMMA GEMM
//   stage 4: flash attention: S^T = K·Q^T, online softmax, acc += V·P^T,
//            ping-pong K fragments (no rotation copies), V-frags hoisted
// ---------------------------------------------------------------------------

typedef __attribute__((ext_vector_type(16))) _Float16 v16h;
typedef __attribute__((ext_vector_type(8)))  float    v8f;

#define WMMA_F16(A, B, Cc) \
  __builtin_amdgcn_wmma_f32_16x16x32_f16(false, (A), false, (B), (short)0, (Cc), false, false)

constexpr int BB = 4;     // batch
constexpr int CC = 512;   // channels
constexpr int CR = 64;    // reduced channels (Cr)
constexpr int NN = 4096;  // H*W
constexpr int KROWS = 2 * CR + CC;  // 640 projection output rows

// ---------------------------------------------------------------------------
// Stage 1: pack the three weight matrices / biases into one f16 GEMM operand.
// ---------------------------------------------------------------------------
__global__ __launch_bounds__(256) void pam_prep_w(
    const float* __restrict__ wb, const float* __restrict__ bb,
    const float* __restrict__ wc, const float* __restrict__ bc,
    const float* __restrict__ wd, const float* __restrict__ bd,
    _Float16* __restrict__ w16, float* __restrict__ bias)
{
  const int i = blockIdx.x * 256 + threadIdx.x;
  if (i < KROWS * CC) {
    const int k = i / CC, c = i % CC;
    float v;
    if (k < CR)           v = wb[(size_t)k * CC + c];
    else if (k < 2 * CR)  v = wc[(size_t)(k - CR) * CC + c];
    else                  v = wd[(size_t)(k - 2 * CR) * CC + c];
    w16[i] = (_Float16)v;
  }
  if (i < KROWS)
    bias[i] = (i < CR) ? bb[i] : (i < 2 * CR) ? bc[i - CR] : bd[i - 2 * CR];
}

// ---------------------------------------------------------------------------
// Stage 2: x[B,C,N] f32 -> xt[B,N,C] f16, 32x32 LDS tiles.
// Global->LDS staging via the CDNA5 async copy (ASYNCcnt-tracked):
// each lane copies one dword from its global address to its LDS address; per-lane LDS byte address is the flat
// shared pointer truncated to 32 bits (ISA 10.2).
// ---------------------------------------------------------------------------
__global__ __launch_bounds__(256) void pam_transpose_x(
    const float* __restrict__ x, _Float16* __restrict__ xt)
{
  __shared__ float tile[32][33];           // +1 pad: conflict-free transpose
  const int n0 = blockIdx.x * 32;
  const int c0 = blockIdx.y * 32;
  const int b  = blockIdx.z;
  const int tn = threadIdx.x & 31;
  const int tr = threadIdx.x >> 5;         // 0..7
#pragma unroll
  for (int j = 0; j < 4; ++j) {
    const int c = tr + j * 8;
    const unsigned ldsoff = (unsigned)(uintptr_t)&tile[c][tn];
    const float* gp = x + ((size_t)b * CC + c0 + c) * NN + n0 + tn;
    asm volatile("global_load_async_to_lds_b32 %0, %1, off"
                 :: "v"(ldsoff), "v"(gp) : "memory");
  }
  asm volatile("s_wait_asynccnt 0" ::: "memory");
  __syncthreads();
#pragma unroll
  for (int j = 0; j < 4; ++j) {
    const int n = tr + j * 8;
    xt[((size_t)b * NN + n0 + n) * CC + c0 + tn] = (_Float16)tile[tn][n];
  }
}

// ---------------------------------------------------------------------------
// Stage 3: projection GEMM on the WMMA pipe.
//   out[k, n] = sum_c w16[k, c] * xt[b, n, c] + bias[k]
// One wave = two 16-row k-tiles x four 16-col n-tiles. The K loop is fully
// unrolled so the double-buffer rotation becomes SSA renames (no v_movs);
// next-step fragments are issued before current-step WMMAs.
// ---------------------------------------------------------------------------
__global__ __launch_bounds__(256) void pam_project_wmma(
    const _Float16* __restrict__ w16, const float* __restrict__ bias,
    const _Float16* __restrict__ xt,
    _Float16* __restrict__ fb, _Float16* __restrict__ fc, _Float16* __restrict__ fd)
{
  const int lane  = threadIdx.x & 31;
  const int wave  = threadIdx.x >> 5;        // 0..7
  const int lmod  = lane & 15;
  const int lhi   = lane >> 4;
  const int k0    = blockIdx.x * 32;         // pair of k-tiles
  const int nbase = (blockIdx.y * 8 + wave) * 64;  // 4 n-tiles of 16
  const int b     = blockIdx.z;

  const _Float16* __restrict__ arow0 = w16 + (size_t)(k0 + lmod) * CC + lhi * 16;
  const _Float16* __restrict__ arow1 = arow0 + (size_t)16 * CC;
  const _Float16* __restrict__ brow =
      xt + ((size_t)b * NN + nbase + lmod) * CC + lhi * 16;

  v8f acc[8] = {};
  v16h a0 = *(const v16h*)(arow0);
  v16h a1 = *(const v16h*)(arow1);
  v16h bt[4];
#pragma unroll
  for (int t = 0; t < 4; ++t) bt[t] = *(const v16h*)(brow + (size_t)(t * 16) * CC);

#pragma unroll
  for (int c = 32; c < CC; c += 32) {
    const v16h an0 = *(const v16h*)(arow0 + c);
    const v16h an1 = *(const v16h*)(arow1 + c);
    v16h bn[4];
#pragma unroll
    for (int t = 0; t < 4; ++t)
      bn[t] = *(const v16h*)(brow + (size_t)(t * 16) * CC + c);
#pragma unroll
    for (int t = 0; t < 4; ++t) {
      acc[t]     = WMMA_F16(a0, bt[t], acc[t]);
      acc[4 + t] = WMMA_F16(a1, bt[t], acc[4 + t]);
    }
    a0 = an0; a1 = an1;                      // SSA renames under full unroll
#pragma unroll
    for (int t = 0; t < 4; ++t) bt[t] = bn[t];
  }
#pragma unroll
  for (int t = 0; t < 4; ++t) {
    acc[t]     = WMMA_F16(a0, bt[t], acc[t]);
    acc[4 + t] = WMMA_F16(a1, bt[t], acc[4 + t]);
  }

  // Epilogue: D layout row = r + 8*lhi, col = lmod.
#pragma unroll
  for (int kt = 0; kt < 2; ++kt) {
    const int ktile = k0 + kt * 16;          // never straddles 64/128 bounds
#pragma unroll
    for (int t = 0; t < 4; ++t) {
      const int n = nbase + t * 16 + lmod;
#pragma unroll
      for (int r = 0; r < 8; ++r) {
        const int k = ktile + lhi * 8 + r;
        const float v = acc[kt * 4 + t][r] + bias[k];
        if (ktile < CR)
          fb[((size_t)b * NN + n) * CR + k] = (_Float16)v;
        else if (ktile < 2 * CR)
          fc[((size_t)b * NN + n) * CR + (k - CR)] = (_Float16)v;
        else
          fd[((size_t)b * CC + (k - 2 * CR)) * NN + n] = (_Float16)v;
      }
    }
  }
}

// ---------------------------------------------------------------------------
// Stage 4: fused attention. Grid = (N/16 query tiles, B); 8 waves, wave owns
// 64 value channels. Key loop unrolled x2 with ping-pong K-fragment buffers
// (zero rotation copies); V-frags issued before the S^T WMMAs; next block's
// K-frags in flight under the softmax VALU/transcendental work.
// ---------------------------------------------------------------------------
__global__ __launch_bounds__(256) void pam_flash_attend(
    const _Float16* __restrict__ fb, const _Float16* __restrict__ fc,
    const _Float16* __restrict__ fd, const float* __restrict__ x,
    const float* __restrict__ alpha, float* __restrict__ out)
{
  const int lane  = threadIdx.x & 31;
  const int wave  = threadIdx.x >> 5;
  const int lmod  = lane & 15;
  const int lhi   = lane >> 4;
  const int qbase = blockIdx.x * 16;
  const int b     = blockIdx.y;
  const int cbase = wave * 64;
  const size_t bN = (size_t)b * NN;

  const _Float16* __restrict__ qrow = fb + (bN + qbase + lmod) * CR + lhi * 16;
  const v16h qB0 = *(const v16h*)(qrow);
  const v16h qB1 = *(const v16h*)(qrow + 32);

  const _Float16* __restrict__ kbase = fc + (bN + lmod) * CR + lhi * 16;
  const _Float16* __restrict__ vbase =
      fd + ((size_t)b * CC + cbase + lmod) * NN + (size_t)lhi * 16;

  v8f acc[4] = {};
  float mrow = -1e30f, lrow = 0.0f;
  const bool lo16 = (lane < 16);

  // One 32-key block: consume kin, load kout for keys (kb+32), update state.
  auto kblock = [&](const v16h (&kin)[4], v16h (&kout)[4], int kb)
      __attribute__((always_inline)) {
    // current block's V fragments — issued before anything consumes them
    v16h vf[4];
#pragma unroll
    for (int t = 0; t < 4; ++t)
      vf[t] = *(const v16h*)(vbase + (size_t)(t * 16) * NN + kb);

    // next block's K fragments (wrap to 0 on the final block; harmless)
    const int knext = (kb + 32 < NN) ? (kb + 32) : 0;
    kout[0] = *(const v16h*)(kbase + (size_t)knext * CR);
    kout[1] = *(const v16h*)(kbase + (size_t)knext * CR + 32);
    kout[2] = *(const v16h*)(kbase + (size_t)(knext + 16) * CR);
    kout[3] = *(const v16h*)(kbase + (size_t)(knext + 16) * CR + 32);
    if (kb + 32 < NN)
      __builtin_prefetch(vbase + kb + 32, 0, 0);   // global_prefetch_b8

    // ---- S^T tiles: 16 keys x 16 queries, reduce over Cr=64
    v8f st0 = {}, st1 = {};
    st0 = WMMA_F16(kin[0], qB0, st0);
    st0 = WMMA_F16(kin[1], qB1, st0);
    st1 = WMMA_F16(kin[2], qB0, st1);
    st1 = WMMA_F16(kin[3], qB1, st1);

    // ---- online softmax (lane = query; keys in VGPR rows + half-wave)
    float smax = -1e30f;
#pragma unroll
    for (int r = 0; r < 8; ++r) smax = fmaxf(smax, fmaxf(st0[r], st1[r]));
    smax = fmaxf(smax, __shfl_xor(smax, 16));
    const float mnew    = fmaxf(mrow, smax);
    const float rescale = __expf(mrow - mnew);

    float p0[8], p1[8], bsum = 0.0f;
#pragma unroll
    for (int r = 0; r < 8; ++r) {
      p0[r] = __expf(st0[r] - mnew);
      p1[r] = __expf(st1[r] - mnew);
      bsum += p0[r] + p1[r];
    }
    bsum += __shfl_xor(bsum, 16);
    lrow = lrow * rescale + bsum;
    mrow = mnew;

    // ---- repack P^T (C/D layout) into a 32x16 B-fragment
    v16h pB{};
#pragma unroll
    for (int r = 0; r < 8; ++r) {
      const float o0 = __shfl_xor(p0[r], 16);
      const float o1 = __shfl_xor(p1[r], 16);
      pB[r]     = (_Float16)(lo16 ? p0[r] : o1);
      pB[r + 8] = (_Float16)(lo16 ? o0 : p1[r]);
    }

    // ---- value accumulation: acc[c,q] += V(16c x 32keys) · P^T
#pragma unroll
    for (int t = 0; t < 4; ++t) {
#pragma unroll
      for (int r = 0; r < 8; ++r) acc[t][r] *= rescale;
      acc[t] = WMMA_F16(vf[t], pB, acc[t]);
    }
  };

  // preload K fragments for block 0, then ping-pong (no rotation copies)
  v16h kfA[4], kfB[4];
  kfA[0] = *(const v16h*)(kbase);
  kfA[1] = *(const v16h*)(kbase + 32);
  kfA[2] = *(const v16h*)(kbase + (size_t)16 * CR);
  kfA[3] = *(const v16h*)(kbase + (size_t)16 * CR + 32);

  for (int kb = 0; kb < NN; kb += 64) {
    kblock(kfA, kfB, kb);
    kblock(kfB, kfA, kb + 32);
  }

  const float inv = (lrow > 0.0f) ? 1.0f / lrow : 0.0f;
  const float alp = alpha[0];
#pragma unroll
  for (int t = 0; t < 4; ++t) {
    const int ch = cbase + t * 16 + lhi * 8;
#pragma unroll
    for (int r = 0; r < 8; ++r) {
      const size_t idx = ((size_t)b * CC + ch + r) * NN + qbase + lmod;
      out[idx] = alp * (acc[t][r] * inv) + x[idx];
    }
  }
}

// ---------------------------------------------------------------------------
extern "C" void kernel_launch(void* const* d_in, const int* in_sizes, int n_in,
                              void* d_out, int out_size, void* d_ws, size_t ws_size,
                              hipStream_t stream)
{
  (void)in_sizes; (void)n_in; (void)out_size; (void)ws_size;
  const float* x     = (const float*)d_in[0];
  const float* wb    = (const float*)d_in[1];
  const float* bb    = (const float*)d_in[2];
  const float* wc    = (const float*)d_in[3];
  const float* bc    = (const float*)d_in[4];
  const float* wd    = (const float*)d_in[5];
  const float* bd    = (const float*)d_in[6];
  const float* alpha = (const float*)d_in[7];
  float* out = (float*)d_out;

  // Workspace layout: fb 2MB, fc 2MB, fd 16MB, xt 16MB, w16 640KB, bias.
  _Float16* fb  = (_Float16*)d_ws;
  _Float16* fc  = fb  + (size_t)BB * NN * CR;
  _Float16* fd  = fc  + (size_t)BB * NN * CR;
  _Float16* xt  = fd  + (size_t)BB * CC * NN;
  _Float16* w16 = xt  + (size_t)BB * NN * CC;
  float*    bias = (float*)(w16 + (size_t)KROWS * CC);

  pam_prep_w<<<dim3((KROWS * CC + 255) / 256), 256, 0, stream>>>(
      wb, bb, wc, bc, wd, bd, w16, bias);

  pam_transpose_x<<<dim3(NN / 32, CC / 32, BB), 256, 0, stream>>>(x, xt);

  pam_project_wmma<<<dim3(KROWS / 32, NN / 512, BB), 256, 0, stream>>>(
      w16, bias, xt, fb, fc, fd);

  pam_flash_attend<<<dim3(NN / 16, BB), 256, 0, stream>>>(fb, fc, fd, x, alpha, out);
}
